// BertSelfAttention_43267500540531
// MI455X (gfx1250) — compile-verified, matrix operand order checked by reading, not run on hardware
//
#include <hip/hip_runtime.h>
#include <hip/hip_bf16.h>

// ---------------------------------------------------------------------------
// BERT self-attention with relative position bias, MI455X (gfx1250, wave32).
// GEMM work on v_wmma_f32_16x16x32_bf16; V staging via Tensor Data Mover.
// ---------------------------------------------------------------------------

typedef __bf16 bf16_t;
typedef __attribute__((ext_vector_type(16))) __bf16 v16bf;
typedef __attribute__((ext_vector_type(8)))  float  v8f;
typedef __attribute__((ext_vector_type(4)))  unsigned int v4u;
typedef __attribute__((ext_vector_type(8)))  int  v8i;
typedef __attribute__((ext_vector_type(4)))  int  v4i;

#define NSEQ  2048
#define HIDD  1024
#define NHEAD 16
#define DHEAD 64
#define WKREL 64   // 2*WK+1 = 129 slots
#define NSLOT 129

// ---- WMMA fragment layout (ISA 7.12.2, wave32) ----------------------------
// 16-bit A 16x32: lane L(0..15) row M=L, lanes 16..31 same rows K+8;
// VGPR g<4 -> K=2g,2g+1 ; g>=4 -> K=16+2(g-4)  (+8 when lane>=16).
__device__ __forceinline__ int frag_kbase(int g, int lane) {
  return ((g < 4) ? 2 * g : 16 + 2 * (g - 4)) + ((lane & 16) ? 8 : 0);
}

// Linear index (lane*16 + elem) of element (r,k) of a 16x32 16-bit fragment
// tile stored in "fragment order" (each lane's 16 values contiguous).
__device__ __forceinline__ int frag_index(int r, int k) {
  int half = (k >> 3) & 1;                 // selects lane group 0-15 / 16-31
  int lane = r + (half << 4);
  int kk = k - (half << 3);                // kk in {0..7} U {16..23}
  int g = (kk < 8) ? (kk >> 1) : (4 + ((kk - 16) >> 1));
  return lane * 16 + 2 * g + (kk & 1);
}

// Fragment load from a fragment-ordered LDS tile: two ds_load_b128.
__device__ __forceinline__ v16bf load_frag_tile(const bf16_t* tile, int lane) {
  return *(const v16bf*)(tile + lane * 16);
}

// A fragment (or B when source stored [n][k] row-major), matrix-order source.
__device__ __forceinline__ v16bf load_frag_rowmajor_bf16(const bf16_t* base, int ld, int lane) {
  v16bf f;
  const bf16_t* p = base + (lane & 15) * ld;
#pragma unroll
  for (int g = 0; g < 8; ++g) {
    int kb = frag_kbase(g, lane);
    f[2 * g]     = p[kb];
    f[2 * g + 1] = p[kb + 1];
  }
  return f;
}

// Same, f32 source (probabilities in LDS) converted on the fly.
__device__ __forceinline__ v16bf load_frag_rowmajor_f32(const float* base, int ld, int lane) {
  v16bf f;
  const float* p = base + (lane & 15) * ld;
#pragma unroll
  for (int g = 0; g < 8; ++g) {
    int kb = frag_kbase(g, lane);
    f[2 * g]     = (bf16_t)p[kb];
    f[2 * g + 1] = (bf16_t)p[kb + 1];
  }
  return f;
}

// B fragment when source is [k][n] (k-major).
__device__ __forceinline__ v16bf load_frag_kmajor_bf16(const bf16_t* base, int ld, int lane) {
  v16bf f;
  int n = lane & 15;
#pragma unroll
  for (int g = 0; g < 8; ++g) {
    int kb = frag_kbase(g, lane);
    f[2 * g]     = base[kb * ld + n];
    f[2 * g + 1] = base[(kb + 1) * ld + n];
  }
  return f;
}

__device__ __forceinline__ v8f wmma_bf16(v16bf a, v16bf b, v8f c) {
  return __builtin_amdgcn_wmma_f32_16x16x32_bf16(false, a, false, b, (short)0, c,
                                                 false, false);
}

// ---- Tensor Data Mover: 2D bf16 tile (32 rows x 64 cols) global -> LDS ----
// D# per CDNA5 ISA (group0 128b, group1 256b, groups2/3 off).
// 6-arg builtin form (clang-23 / therock-10.0 headers).
__device__ __forceinline__ void tdm_load_v_chunk(const bf16_t* gsrc, unsigned lds_off) {
  unsigned long long ga = (unsigned long long)(uintptr_t)gsrc;
  v4u g0 = { 1u,                                   // count=1, user mode
             lds_off,                              // lds_addr (bytes)
             (unsigned)(ga & 0xffffffffu),         // global_addr[31:0]
             (unsigned)((ga >> 32) & 0x01ffffffu)  // global_addr[56:32]
               | (2u << 30) };                     // type=2 ("image")
  v8i g1 = { (int)(1u << 16),      // data_size=1 (2 bytes); wg_mask=0
             (int)((unsigned)DHEAD << 16),  // tensor_dim0=64 (bits 79:48 lo)
             (int)((unsigned)NSEQ  << 16),  // tensor_dim1=2048 (bits 111:80 lo)
             (int)((unsigned)DHEAD << 16),  // tile_dim0=64 (bits 127:112)
             32,                   // tile_dim1=32, tile_dim2=0
             DHEAD,                // tensor_dim0_stride=64 (bits 207:160 lo)
             0, 0 };               // dim1 stride unused (2D)
  v4i z4 = {0, 0, 0, 0};
  v8i z8 = {0, 0, 0, 0, 0, 0, 0, 0};
  __builtin_amdgcn_tensor_load_to_lds(g0, g1, z4, z4, z8, 0);
}

// ---------------------------------------------------------------------------
// Kernel 1: fused QKV projection -> bf16 per-head [H][N][DH].
// grid = (32 row tiles, 16 heads, 3 matrices), 256 threads (8 waves).
// LDS tiles stored in WMMA fragment order: fragment load = 2x ds_load_b128.
// ---------------------------------------------------------------------------
__global__ __launch_bounds__(256) void qkv_proj_kernel(
    const float* __restrict__ X,
    const float* __restrict__ Wq, const float* __restrict__ bq,
    const float* __restrict__ Wk, const float* __restrict__ bk,
    const float* __restrict__ Wv, const float* __restrict__ bv,
    bf16_t* __restrict__ Qo, bf16_t* __restrict__ Ko, bf16_t* __restrict__ Vo) {
  const int sel = blockIdx.z;
  const float* W  = (sel == 0) ? Wq : (sel == 1) ? Wk : Wv;
  const float* bb = (sel == 0) ? bq : (sel == 1) ? bk : bv;
  bf16_t* O       = (sel == 0) ? Qo : (sel == 1) ? Ko : Vo;

  const int rowBase = blockIdx.x * 64;
  const int head    = blockIdx.y;
  const int colBase = head * 64;

  // 64x64 block = 4x2 fragment tiles (16x32 each), fragment-ordered.
  __shared__ alignas(32) bf16_t XsT[8][512];   // [mtile*2 + ktile][frag idx]
  __shared__ alignas(32) bf16_t WsT[8][512];   // [ntile*2 + ktile], src [n][k]

  const int tid  = threadIdx.x;
  const int lane = tid & 31;
  const int wave = tid >> 5;
  const int mt   = wave >> 1;                  // wave's row tile (0..3)

  v8f acc[2] = {v8f{}, v8f{}};

  for (int kk = 0; kk < HIDD; kk += 64) {
    for (int e = tid; e < 64 * 64; e += 256) {
      int m = e >> 6, c = e & 63;
      XsT[((m >> 4) << 1) + (c >> 5)][frag_index(m & 15, c & 31)] =
          (bf16_t)X[(size_t)(rowBase + m) * HIDD + kk + c];
    }
    for (int e = tid; e < 64 * 64; e += 256) {
      int k = e >> 6, n = e & 63;
      WsT[((n >> 4) << 1) + (k >> 5)][frag_index(n & 15, k & 31)] =
          (bf16_t)W[(size_t)(kk + k) * HIDD + colBase + n];
    }
    __syncthreads();

    v16bf a0 = load_frag_tile(XsT[mt * 2 + 0], lane);
    v16bf a1 = load_frag_tile(XsT[mt * 2 + 1], lane);
#pragma unroll
    for (int i = 0; i < 2; ++i) {
      int nt = (wave * 2 + i) & 3;
      v16bf b0 = load_frag_tile(WsT[nt * 2 + 0], lane);
      v16bf b1 = load_frag_tile(WsT[nt * 2 + 1], lane);
      acc[i] = wmma_bf16(a0, b0, acc[i]);
      acc[i] = wmma_bf16(a1, b1, acc[i]);
    }
    __syncthreads();
  }

  const int n  = lane & 15;
  const int mo = (lane & 16) ? 8 : 0;
#pragma unroll
  for (int i = 0; i < 2; ++i) {
    int t = wave * 2 + i;
    int mrow = (t >> 2) * 16, ncol = (t & 3) * 16;
    int d = ncol + n;
    float bias = bb[colBase + d];
#pragma unroll
    for (int r = 0; r < 8; ++r) {
      int row = rowBase + mrow + mo + r;
      O[((size_t)head * NSEQ + row) * DHEAD + d] = (bf16_t)(acc[i][r] + bias);
    }
  }
}

// ---------------------------------------------------------------------------
// Kernel 2: attention for one (head, 32-query-row block).
// 32x2048 f32 score strip in LDS (256KB -> CDNA5 320KB WGP LDS).
// V chunks staged by the Tensor Data Mover, double buffered.
// grid = (64 row blocks, 16 heads), 256 threads (8 waves).
// ---------------------------------------------------------------------------
__global__ __launch_bounds__(256) void attn_kernel(
    const bf16_t* __restrict__ Q, const bf16_t* __restrict__ K,
    const bf16_t* __restrict__ V, const float* __restrict__ mask,
    const float* __restrict__ Wrk,   // [DH][129]
    const float* __restrict__ Wrv,   // [129][DH]
    float* __restrict__ out) {
  __shared__ float S[32][NSEQ];                  // 256 KB score/prob strip
  __shared__ alignas(32) bf16_t QsT[4][512];     // Q fragments: [mtile*2+ktile]
  __shared__ alignas(16) bf16_t Vs[2][32][DHEAD];// TDM double buffer (2x4KB)

  const int head    = blockIdx.y;
  const int rowBase = blockIdx.x * 32;
  const int tid  = threadIdx.x;
  const int lane = tid & 31;
  const int wave = tid >> 5;

  const bf16_t* Qh = Q + (size_t)head * NSEQ * DHEAD;
  const bf16_t* Kh = K + (size_t)head * NSEQ * DHEAD;
  const bf16_t* Vh = V + (size_t)head * NSEQ * DHEAD;

  for (int e = tid; e < 32 * DHEAD; e += 256) {
    int i = e >> 6, d = e & 63;
    QsT[((i >> 4) << 1) + (d >> 5)][frag_index(i & 15, d & 31)] =
        Qh[(size_t)(rowBase + i) * DHEAD + d];
  }
  __syncthreads();

  // -------- Phase A: S = Q Khᵀ; Q fragments hoisted out of the loop --------
  {
    const v16bf a00 = load_frag_tile(QsT[0], lane);  // rows 0..15 , k 0..31
    const v16bf a01 = load_frag_tile(QsT[1], lane);  // rows 0..15 , k 32..63
    const v16bf a10 = load_frag_tile(QsT[2], lane);  // rows 16..31, k 0..31
    const v16bf a11 = load_frag_tile(QsT[3], lane);  // rows 16..31, k 32..63
    for (int ct = wave; ct < NSEQ / 16; ct += 8) {   // wave-uniform
      int j0 = ct * 16;
      if (ct + 8 < NSEQ / 16)
        __builtin_prefetch(&Kh[(size_t)(j0 + 128) * DHEAD], 0, 1);
      v8f c0 = {}, c1 = {};
      v16bf b0 = load_frag_rowmajor_bf16(&Kh[(size_t)j0 * DHEAD], DHEAD, lane);
      v16bf b1 = load_frag_rowmajor_bf16(&Kh[(size_t)j0 * DHEAD + 32], DHEAD, lane);
      c0 = wmma_bf16(a00, b0, c0);
      c0 = wmma_bf16(a01, b1, c0);
      c1 = wmma_bf16(a10, b0, c1);
      c1 = wmma_bf16(a11, b1, c1);
      int n = lane & 15, mo = (lane & 16) ? 8 : 0;
#pragma unroll
      for (int r = 0; r < 8; ++r) {
        S[mo + r][j0 + n]      = c0[r];
        S[16 + mo + r][j0 + n] = c1[r];
      }
    }
  }
  __syncthreads();

  // -------- Phase B1: relative-key bias on the diagonal band ---------------
  for (int e = tid; e < 32 * NSLOT; e += 256) {
    int i = e / NSLOT, slot = e % NSLOT;
    int j = rowBase + i - WKREL + slot;
    if (j >= 0 && j < NSEQ) {
      float dot = 0.f;
      int tbase = ((i >> 4) << 1);
      int rl = i & 15;
#pragma unroll 8
      for (int d = 0; d < DHEAD; ++d)
        dot += (float)QsT[tbase + (d >> 5)][frag_index(rl, d & 31)] *
               Wrk[d * NSLOT + slot];
      S[i][j] += dot;
    }
  }
  __syncthreads();

  // -------- Phase B2: scale (1/sqrt(64)) + additive mask -------------------
  for (int e = tid; e < 32 * NSEQ; e += 256) {
    int i = e >> 11, j = e & (NSEQ - 1);
    float ext = (1.0f - mask[j]) * -3.4028234663852886e38f;
    S[i][j] = S[i][j] * 0.125f + ext;
  }
  __syncthreads();

  // -------- Phase C: row softmax (wave-parallel, wave32 shuffles) ----------
  for (int r = wave * 4; r < wave * 4 + 4; ++r) {
    float m = -3.4028234663852886e38f;
    for (int j = lane; j < NSEQ; j += 32) m = fmaxf(m, S[r][j]);
#pragma unroll
    for (int off = 16; off; off >>= 1) m = fmaxf(m, __shfl_xor(m, off, 32));
    float sum = 0.f;
    for (int j = lane; j < NSEQ; j += 32) {
      float p = __expf(S[r][j] - m);
      S[r][j] = p;
      sum += p;
    }
#pragma unroll
    for (int off = 16; off; off >>= 1) sum += __shfl_xor(sum, off, 32);
    float inv = 1.0f / sum;
    for (int j = lane; j < NSEQ; j += 32) S[r][j] *= inv;
  }
  __syncthreads();

  // -------- Phase D: ctx = P V, TDM double-buffered V chunks ---------------
  // wave w owns ctx tile mrow=(w&1)*16, ncol=(w>>1)*16 of the 32x64 block.
  const int mrow = (wave & 1) * 16;
  const int ncol = (wave >> 1) * 16;
  if (wave == 0) {                                  // wave-uniform branch
    tdm_load_v_chunk(Vh, (unsigned)(uintptr_t)&Vs[0][0][0]);
    __builtin_amdgcn_s_wait_tensorcnt(0);
  }
  __syncthreads();

  v8f acc = {};
  int cur = 0;
  for (int jt = 0; jt < NSEQ; jt += 32) {
    if (wave == 0 && jt + 32 < NSEQ)               // prefetch next chunk
      tdm_load_v_chunk(Vh + (size_t)(jt + 32) * DHEAD,
                       (unsigned)(uintptr_t)&Vs[cur ^ 1][0][0]);
    v16bf a = load_frag_rowmajor_f32(&S[mrow][jt], NSEQ, lane);
    v16bf b = load_frag_kmajor_bf16(&Vs[cur][0][ncol], DHEAD, lane);
    acc = wmma_bf16(a, b, acc);
    if (wave == 0) __builtin_amdgcn_s_wait_tensorcnt(0);
    __syncthreads();
    cur ^= 1;
  }

  // -------- Epilogue: band @ W_rel_v and writeout --------------------------
  const int n  = lane & 15;
  const int mo = (lane & 16) ? 8 : 0;
#pragma unroll
  for (int r = 0; r < 8; ++r) {
    int i   = mrow + mo + r;
    int row = rowBase + i;
    int d   = ncol + n;
    float extra = 0.f;
    for (int slot = 0; slot < NSLOT; ++slot) {
      int kpos = row - WKREL + slot;
      if (kpos >= 0 && kpos < NSEQ) extra += S[i][kpos] * Wrv[slot * DHEAD + d];
    }
    out[(size_t)row * HIDD + head * DHEAD + d] = acc[r] + extra;
  }
}

// ---------------------------------------------------------------------------
extern "C" void kernel_launch(void* const* d_in, const int* in_sizes, int n_in,
                              void* d_out, int out_size, void* d_ws, size_t ws_size,
                              hipStream_t stream) {
  const float* hidden = (const float*)d_in[0];
  const float* mask   = (const float*)d_in[1];
  const float* Wq     = (const float*)d_in[2];
  const float* bq     = (const float*)d_in[3];
  const float* Wk     = (const float*)d_in[4];
  const float* bk     = (const float*)d_in[5];
  const float* Wv     = (const float*)d_in[6];
  const float* bv     = (const float*)d_in[7];
  const float* Wrk    = (const float*)d_in[8];
  const float* Wrv    = (const float*)d_in[9];
  float* out = (float*)d_out;

  const size_t qbytes = (size_t)NHEAD * NSEQ * DHEAD * sizeof(bf16_t);
  bf16_t* Qb = (bf16_t*)d_ws;
  bf16_t* Kb = (bf16_t*)((char*)d_ws + qbytes);
  bf16_t* Vb = (bf16_t*)((char*)d_ws + 2 * qbytes);

  qkv_proj_kernel<<<dim3(NSEQ / 64, NHEAD, 3), 256, 0, stream>>>(
      hidden, Wq, bq, Wk, bk, Wv, bv, Qb, Kb, Vb);

  attn_kernel<<<dim3(NSEQ / 32, NHEAD), 256, 0, stream>>>(
      Qb, Kb, Vb, mask, Wrk, Wrv, out);
}
